// SequenceEncoder_17789754540972
// MI455X (gfx1250) — matrix operand order repaired and checked by reading, not compiled, not run on hardware
//
#include <hip/hip_runtime.h>
#include <hip/hip_bf16.h>

typedef __attribute__((ext_vector_type(16))) _Float16 v16h;
typedef __attribute__((ext_vector_type(8)))  _Float16 v8h;
typedef __attribute__((ext_vector_type(8)))  float    v8f;
typedef __attribute__((ext_vector_type(4)))  float    v4f;

#define EDIM 64
#define HDIM 64
#define TSEQ 50
#define NITEMS 100000
#define GSTRIDE 260   // gates LDS row stride in floats: 8*260 % 64 == 32 -> conflict-free half-wave writes

__device__ __forceinline__ float fast_rcp(float x) {
  return __builtin_amdgcn_rcpf(x);        // single v_rcp_f32 (TRANS pipe)
}
__device__ __forceinline__ float fast_sigmoid(float x) {
  return fast_rcp(1.0f + __expf(-x));
}
__device__ __forceinline__ float fast_tanh(float x) {
  // 1 - 2/(e^{2x}+1): monotone, no NaN at extremes
  float e = __expf(2.0f * x);
  return 1.0f - 2.0f * fast_rcp(e + 1.0f);
}

__global__ void zero_f32_kernel(float* __restrict__ p, int n) {
  int i = blockIdx.x * blockDim.x + threadIdx.x;
  if (i < n) p[i] = 0.0f;
}

__global__ __launch_bounds__(256) void seq_lstm_kernel(
    const int*   __restrict__ seq,    // [U, T]
    const int*   __restrict__ lens,   // [U]
    const float* __restrict__ emb,    // [NI, E]
    const float* __restrict__ Wih,    // [4H, E]
    const float* __restrict__ Whh,    // [4H, H]
    const float* __restrict__ bih,    // [4H]
    const float* __restrict__ bhh,    // [4H]
    float*       __restrict__ sums,   // [NI, H]  (accumulated with atomics)
    float*       __restrict__ counts) // [NI]
{
  __shared__ __align__(16) _Float16 xlds[16 * EDIM];
  __shared__ __align__(16) _Float16 hlds[16 * HDIM];
  __shared__ __align__(16) float    gates[16 * GSTRIDE];
  __shared__ int idsAll[16 * TSEQ];   // all item ids for this 16-user tile
  __shared__ int lenl[16];

  const int tid  = threadIdx.x;
  const int lane = tid & 31;
  const int wv   = tid >> 5;           // wave 0..7, owns N-tiles 2w,2w+1
  const int ub   = blockIdx.x * 16;    // user tile base

  // h0 = 0
  #pragma unroll
  for (int q = 0; q < 4; ++q) hlds[tid + 256 * q] = (_Float16)0.0f;
  if (tid < 16) lenl[tid] = lens[ub + tid];

  // preload all sequence ids for the tile (rows contiguous -> flat coalesced copy)
  {
    const int* srow = seq + ub * TSEQ;
    for (int i = tid; i < 16 * TSEQ; i += 256) idsAll[i] = srow[i];
  }

  // ---- preload WMMA B fragments (weights, f32 -> f16), resident in VGPRs ----
  // B k order used here (matches A): lane n holds K[0..15] of column n; lane n+16 holds K[16..31].
  // B = W^T so column n of B == row gate_n of W (contiguous in memory).
  const int nloc = lane & 15;
  const int kblk = (lane >> 4) * 16;
  v16h Bih[2][2], Bhh[2][2];
  float bv[2];
  #pragma unroll
  for (int tt = 0; tt < 2; ++tt) {
    const int n = (wv * 2 + tt) * 16 + nloc;
    bv[tt] = bih[n] + bhh[n];                    // bias folded into WMMA C
    #pragma unroll
    for (int kf = 0; kf < 2; ++kf) {
      const float* pi = Wih + n * EDIM + kf * 32 + kblk;
      const float* ph = Whh + n * HDIM + kf * 32 + kblk;
      v16h bi, bh;
      #pragma unroll
      for (int c = 0; c < 4; ++c) {
        v4f vi = *(const v4f*)(pi + c * 4);
        v4f vh = *(const v4f*)(ph + c * 4);
        bi[c * 4 + 0] = (_Float16)vi.x; bi[c * 4 + 1] = (_Float16)vi.y;
        bi[c * 4 + 2] = (_Float16)vi.z; bi[c * 4 + 3] = (_Float16)vi.w;
        bh[c * 4 + 0] = (_Float16)vh.x; bh[c * 4 + 1] = (_Float16)vh.y;
        bh[c * 4 + 2] = (_Float16)vh.z; bh[c * 4 + 3] = (_Float16)vh.w;
      }
      Bih[tt][kf] = bi;
      Bhh[tt][kf] = bh;
    }
  }

  // elementwise thread map: pair p = tid + 256*q -> (m = p>>6, j = p&63); c stays in regs
  float cst[4] = {0.0f, 0.0f, 0.0f, 0.0f};
  const int ju = tid & 63;
  const int mb = tid >> 6;
  // gather map: 16 threads per user row of 64 floats
  const int gu = tid >> 4;
  const int ge = (tid & 15) * 4;
  // A fragment map (16-bit A 16x32): lane m -> K{0..15}; lane m+16 -> K{16..31}
  const int am = lane & 15;
  const int ao = (lane >> 4) * 16;

  __syncthreads();   // idsAll visible before first prefetch

  // prefetch embedding row chunk for t = 0
  v4f vx = *(const v4f*)(emb + (size_t)idsAll[gu * TSEQ] * EDIM + ge);

  for (int t = 0; t < TSEQ; ++t) {
    // ---- commit prefetched x_t (f32 -> f16 into LDS) ----
    _Float16* xp = xlds + gu * EDIM + ge;
    xp[0] = (_Float16)vx.x; xp[1] = (_Float16)vx.y;
    xp[2] = (_Float16)vx.z; xp[3] = (_Float16)vx.w;
    __syncthreads();

    // ---- issue prefetch for t+1 (clamped; latency hidden behind WMMA + cell) ----
    {
      const int tn = (t + 1 < TSEQ) ? (t + 1) : (TSEQ - 1);
      vx = *(const v4f*)(emb + (size_t)idsAll[gu * TSEQ + tn] * EDIM + ge);
    }

    // ---- A fragments for x_t and h_{t-1} ----
    v16h ax[2], ah[2];
    #pragma unroll
    for (int kf = 0; kf < 2; ++kf) {
      const _Float16* xr = xlds + am * EDIM + kf * 32;
      const _Float16* hr = hlds + am * HDIM + kf * 32;
      v8h xlo = *(const v8h*)(xr + ao);
      v8h xhi = *(const v8h*)(xr + ao + 8);
      v8h hlo = *(const v8h*)(hr + ao);
      v8h hhi = *(const v8h*)(hr + ao + 8);
      ax[kf] = __builtin_shufflevector(xlo, xhi, 0,1,2,3,4,5,6,7,8,9,10,11,12,13,14,15);
      ah[kf] = __builtin_shufflevector(hlo, hhi, 0,1,2,3,4,5,6,7,8,9,10,11,12,13,14,15);
    }

    // ---- gates tile = x W_ih^T + h W_hh^T + bias, via chained WMMA f32<-f16 ----
    #pragma unroll
    for (int tt = 0; tt < 2; ++tt) {
      v8f acc = { bv[tt], bv[tt], bv[tt], bv[tt], bv[tt], bv[tt], bv[tt], bv[tt] };
      acc = __builtin_amdgcn_wmma_f32_16x16x32_f16(false, ax[0], false, Bih[tt][0], (short)0, acc, false, false);
      acc = __builtin_amdgcn_wmma_f32_16x16x32_f16(false, ax[1], false, Bih[tt][1], (short)0, acc, false, false);
      acc = __builtin_amdgcn_wmma_f32_16x16x32_f16(false, ah[0], false, Bhh[tt][0], (short)0, acc, false, false);
      acc = __builtin_amdgcn_wmma_f32_16x16x32_f16(false, ah[1], false, Bhh[tt][1], (short)0, acc, false, false);
      const int nb = (wv * 2 + tt) * 16 + nloc;
      #pragma unroll
      for (int r = 0; r < 8; ++r)   // C/D layout: VGPR r = row (r + 8*(lane/16)), col = lane%16
        gates[((lane >> 4) * 8 + r) * GSTRIDE + nb] = acc[r];
    }
    __syncthreads();

    // ---- elementwise LSTM cell + masked scatter ----
    #pragma unroll
    for (int q = 0; q < 4; ++q) {
      const int m = mb + 4 * q;
      const float* gr = gates + m * GSTRIDE;
      float gi = fast_sigmoid(gr[ju]);
      float gf = fast_sigmoid(gr[64 + ju]);
      float gg = fast_tanh(gr[128 + ju]);
      float go = fast_sigmoid(gr[192 + ju]);
      float c  = gf * cst[q] + gi * gg;
      cst[q] = c;
      float h = go * fast_tanh(c);
      hlds[m * HDIM + ju] = (_Float16)h;
      if (t < lenl[m]) {
        const int item = idsAll[m * TSEQ + t];
        atomicAdd(sums + item * HDIM + ju, h);
        if (ju == 0) atomicAdd(counts + item, 1.0f);
      }
    }
    __syncthreads();
  }
}

__global__ void finalize_kernel(float* __restrict__ out, const float* __restrict__ counts, int n) {
  int i = blockIdx.x * blockDim.x + threadIdx.x;
  if (i < n) {
    float cnt = counts[i >> 6];
    float s = out[i];
    out[i] = (cnt > 0.0f) ? (s * fast_rcp(cnt)) : 0.0f;
  }
}

extern "C" void kernel_launch(void* const* d_in, const int* in_sizes, int n_in,
                              void* d_out, int out_size, void* d_ws, size_t ws_size,
                              hipStream_t stream) {
  const int*   seq  = (const int*)d_in[0];
  const int*   lens = (const int*)d_in[1];
  const float* emb  = (const float*)d_in[2];
  const float* Wih  = (const float*)d_in[3];
  const float* Whh  = (const float*)d_in[4];
  const float* bih  = (const float*)d_in[5];
  const float* bhh  = (const float*)d_in[6];
  float* out    = (float*)d_out;        // doubles as sums accumulator [NI, H]
  float* counts = (float*)d_ws;         // [NI] floats in workspace

  const int nU = in_sizes[0] / TSEQ;    // 16384
  const int outElems = NITEMS * HDIM;   // == out_size

  // zero accumulators (d_out and counts are poisoned by the harness)
  zero_f32_kernel<<<(outElems + 255) / 256, 256, 0, stream>>>(out, outElems);
  zero_f32_kernel<<<(NITEMS + 255) / 256, 256, 0, stream>>>(counts, NITEMS);

  // LSTM over all users: one block per 16-user tile, 8 waves each
  seq_lstm_kernel<<<nU / 16, 256, 0, stream>>>(seq, lens, emb, Wih, Whh, bih, bhh, out, counts);

  // segment mean with zero-fill
  finalize_kernel<<<(outElems + 255) / 256, 256, 0, stream>>>(out, counts, outElems);
}
